// GraphTransformer_42356967473340
// MI455X (gfx1250) — matrix operand order repaired and checked by reading, not compiled, not run on hardware
//
#include <hip/hip_runtime.h>

// ---------------------------------------------------------------------------
// MI455X / gfx1250 fused transformer block.
// WMMA f16 (16x16x32) with fp32 accumulation for every matmul.
// Flash-style online-softmax attention (no materialized W).
// ---------------------------------------------------------------------------

typedef __attribute__((ext_vector_type(16))) _Float16 v16h;
typedef __attribute__((ext_vector_type(8)))  _Float16 v8h;
typedef __attribute__((ext_vector_type(8)))  float    v8f;

#define B_    8
#define L_    896
#define DM_   320
#define NH_   10
#define DH_   32
#define DI_   1280
#define ROWS_ (B_ * L_)          // 7168
#define LN_EPS_ 1e-5f

// ---- A-fragment loader: 16-bit A 16x32, contiguous k order (same as packed B) -----------------
// lane r (both halves) holds row r&15; element i -> k = 2*hi8 + i
__device__ __forceinline__ v16h load_a16(const _Float16* p, int hi8) {
  v8h lo = *(const v8h*)(p + 2 * hi8);
  v8h hi = *(const v8h*)(p + 2 * hi8 + 8);
  v16h r;
#pragma unroll
  for (int i = 0; i < 8; ++i) { r[i] = lo[i]; r[i + 8] = hi[i]; }
  return r;
}

__device__ __forceinline__ v8f wmma16(v16h a, v16h b, v8f c) {
  return __builtin_amdgcn_wmma_f32_16x16x32_f16(false, a, false, b, (short)0, c,
                                                false, false);
}

// ---------------------------------------------------------------------------
// LayerNorm (one wave per 320-wide row) -> f16
// ---------------------------------------------------------------------------
__global__ void __launch_bounds__(256)
ln_f16_kernel(const float* __restrict__ x, const float* __restrict__ g,
              const float* __restrict__ bta, _Float16* __restrict__ out) {
  int row  = blockIdx.x * (blockDim.x >> 5) + (threadIdx.x >> 5);
  int lane = threadIdx.x & 31;
  if (row >= ROWS_) return;
  const float* xr = x + (size_t)row * DM_;
  float v[10];
  float s = 0.f;
#pragma unroll
  for (int i = 0; i < 10; ++i) { v[i] = xr[lane + 32 * i]; s += v[i]; }
#pragma unroll
  for (int o = 16; o > 0; o >>= 1) s += __shfl_xor(s, o, 32);
  float mu = s * (1.0f / DM_);
  float vs = 0.f;
#pragma unroll
  for (int i = 0; i < 10; ++i) { float d = v[i] - mu; vs += d * d; }
#pragma unroll
  for (int o = 16; o > 0; o >>= 1) vs += __shfl_xor(vs, o, 32);
  float rs = rsqrtf(vs * (1.0f / DM_) + LN_EPS_);
  _Float16* orow = out + (size_t)row * DM_;
#pragma unroll
  for (int i = 0; i < 10; ++i) {
    int c = lane + 32 * i;
    orow[c] = (_Float16)((v[i] - mu) * rs * g[c] + bta[c]);
  }
}

// ---------------------------------------------------------------------------
// Pack f32 weight [K,N] into f16 B-fragment layout:
// frag t=(kt*NT+nt): element (lane,i) = w[kt*32 + 16*(lane>=16)+i, nt*16+(lane&15)]
// ---------------------------------------------------------------------------
__global__ void __launch_bounds__(256)
pack_b_kernel(const float* __restrict__ w, _Float16* __restrict__ pk,
              int K, int N) {
  int idx = blockIdx.x * blockDim.x + threadIdx.x;
  if (idx >= K * N) return;
  int i    = idx & 15;
  int lane = (idx >> 4) & 31;
  int t    = idx >> 9;
  int NT   = N >> 4;
  int nt   = t % NT, kt = t / NT;
  int k = kt * 32 + ((lane >> 4) << 4) + i;
  int n = nt * 16 + (lane & 15);
  pk[idx] = (_Float16)w[(size_t)k * N + n];
}

// ---------------------------------------------------------------------------
// Generic one-wave 16x16 tile GEMM core: C = A[ROWS,K] * Bpacked[K,N]
// ---------------------------------------------------------------------------
__device__ __forceinline__ v8f gemm_tile(const _Float16* __restrict__ A, int lda,
                                         const _Float16* __restrict__ Bpk,
                                         int mt, int nt, int NT, int KT,
                                         int lane) {
  int hi8 = ((lane >> 4) << 3);
  int col = lane & 15;
  const _Float16* arow = A + (size_t)(mt * 16 + col) * lda;
  const _Float16* bp   = Bpk + ((size_t)nt << 9) + ((size_t)lane << 4);
  v8f acc = {};
  for (int kt = 0; kt < KT; ++kt) {
    if (kt + 1 < KT) __builtin_prefetch(arow + (kt + 1) * 32, 0, 1);
    v16h a  = load_a16(arow + kt * 32, hi8);
    v16h bf = *(const v16h*)(bp + (((size_t)kt * NT) << 9));
    acc = wmma16(a, bf, acc);
  }
  return acc;
}

// ---------------------------------------------------------------------------
// QKV gemm: Zn[7168,320] @ wqkv[320,960] + b -> scatter to Qh/Kh/Vt (f16)
//   Qh [B*H, L, 32]  (pre-scaled by 1/sqrt(32))
//   Kh [B*H, L, 32]
//   Vt [B*H, 32, L]  (transposed V for PV B-fragments)
// ---------------------------------------------------------------------------
__global__ void __launch_bounds__(256)
qkv_gemm_kernel(const _Float16* __restrict__ A, const _Float16* __restrict__ Bpk,
                const float* __restrict__ bias, _Float16* __restrict__ Qh,
                _Float16* __restrict__ Kh, _Float16* __restrict__ Vt) {
  const int NT = 60, KT = 10;
  int gw   = blockIdx.x * (blockDim.x >> 5) + (threadIdx.x >> 5);
  int lane = threadIdx.x & 31;
  if (gw >= (ROWS_ / 16) * NT) return;
  int mt = gw / NT, nt = gw % NT;
  v8f acc = gemm_tile(A, DM_, Bpk, mt, nt, NT, KT, lane);
  int hi8 = ((lane >> 4) << 3);
  int col = lane & 15;
  int n0 = nt * 16 + col;
  float bv = bias[n0];
  int h = n0 / 96, wc = n0 % 96;
  const float isd = 0.17677669529663687f;  // 1/sqrt(32)
#pragma unroll
  for (int j = 0; j < 8; ++j) {
    int row = mt * 16 + hi8 + j;
    int b = row / L_, l = row % L_;
    size_t bh = (size_t)b * NH_ + h;
    float v = acc[j] + bv;
    if (wc < 32) {
      Vt[(bh * DH_ + wc) * L_ + l] = (_Float16)v;
    } else if (wc < 64) {
      Qh[(bh * L_ + l) * DH_ + (wc - 32)] = (_Float16)(v * isd);
    } else {
      Kh[(bh * L_ + l) * DH_ + (wc - 64)] = (_Float16)v;
    }
  }
}

// ---------------------------------------------------------------------------
// Fused flash attention: one wave per (b,h,ntile of 16 rows).
// Per 32-wide m-chunk: 2 score WMMAs + online softmax + 2 PV WMMAs.
// Post-softmax mask folded into P; leaky-ReLU epilogue.
// ---------------------------------------------------------------------------
#define SEG(i) ((i) < segN ? 0 : ((i) < segNM ? 1 : 2))

__global__ void __launch_bounds__(128)
attn_kernel(const _Float16* __restrict__ Qh, const _Float16* __restrict__ Kh,
            const _Float16* __restrict__ Vt, const float* __restrict__ Dm,
            const float* __restrict__ nmsk, const float* __restrict__ msk,
            const float* __restrict__ gamma, const int* __restrict__ pN,
            const int* __restrict__ pM, _Float16* __restrict__ attn) {
  __shared__ _Float16 smem[4][16 * 40];  // padded stride: 40 halfs (16B-aligned, bank-safe)
  int wv   = threadIdx.x >> 5;
  int lane = threadIdx.x & 31;
  int gw   = blockIdx.x * 4 + wv;
  const int NTL = L_ / 16;  // 56
  if (gw >= B_ * NH_ * NTL) return;
  int ntile = gw % NTL;
  int bh    = gw / NTL;
  int b = bh / NH_, h = bh % NH_;
  int segN = pN[0], segNM = segN + pM[0];
  int hi  = lane >> 4;
  int hi8 = hi << 3;
  int col = lane & 15;

  // Q A-fragment, loaded once (rows = ntile*16 + (lane&15), k = d)
  v16h aQ = load_a16(Qh + ((size_t)bh * L_ + ntile * 16 + col) * DH_, hi8);

  float rmax[8], rsum[8];
  v8f o0 = {}, o1 = {};
#pragma unroll
  for (int j = 0; j < 8; ++j) { rmax[j] = -3.0e38f; rsum[j] = 0.f; }

  for (int mc = 0; mc < L_ / 32; ++mc) {
    int m0 = mc * 32;
    // --- scores: two 16x16 WMMAs (K = d = 32) ---
    const _Float16* kb0 = Kh + ((size_t)bh * L_ + m0 + col) * DH_ + (hi << 4);
    v16h bK0 = *(const v16h*)kb0;
    v16h bK1 = *(const v16h*)(kb0 + 16 * DH_);
    v8f z0 = {}, z1 = {};
    v8f s0 = wmma16(aQ, bK0, z0);
    v8f s1 = wmma16(aQ, bK1, z1);

    int mA = m0 + col, mB = m0 + 16 + col;
    int smA = SEG(mA), smB = SEG(mB);
    float e0[8], e1[8];
#pragma unroll
    for (int j = 0; j < 8; ++j) {
      int nloc = ntile * 16 + hi8 + j;
      int sn = SEG(nloc);
      size_t ro = ((size_t)b * L_ + nloc) * (size_t)L_;
      e0[j] = s0[j] - gamma[sn * 4 + smA] * Dm[ro + mA] + nmsk[ro + mA];
      e1[j] = s1[j] - gamma[sn * 4 + smB] * Dm[ro + mB] + nmsk[ro + mB];
    }
    // --- online softmax (row stats across the 16-lane half) ---
#pragma unroll
    for (int j = 0; j < 8; ++j) {
      float mx = fmaxf(e0[j], e1[j]);
      mx = fmaxf(mx, __shfl_xor(mx, 1, 32));
      mx = fmaxf(mx, __shfl_xor(mx, 2, 32));
      mx = fmaxf(mx, __shfl_xor(mx, 4, 32));
      mx = fmaxf(mx, __shfl_xor(mx, 8, 32));
      float mn = fmaxf(rmax[j], mx);
      float alpha = __expf(rmax[j] - mn);
      rmax[j] = mn;
      float p0 = __expf(e0[j] - mn);
      float p1 = __expf(e1[j] - mn);
      float sr = p0 + p1;
      sr += __shfl_xor(sr, 1, 32);
      sr += __shfl_xor(sr, 2, 32);
      sr += __shfl_xor(sr, 4, 32);
      sr += __shfl_xor(sr, 8, 32);
      rsum[j] = rsum[j] * alpha + sr;
      o0[j] *= alpha;
      o1[j] *= alpha;
      e0[j] = p0;
      e1[j] = p1;
    }
    // --- P*mask -> LDS (C-layout -> A-layout bounce) ---
    _Float16* pb = &smem[wv][0];
#pragma unroll
    for (int j = 0; j < 8; ++j) {
      int nl = hi8 + j;
      size_t ro = ((size_t)b * L_ + (ntile * 16 + nl)) * (size_t)L_;
      pb[nl * 40 + col]      = (_Float16)(e0[j] * msk[ro + mA]);
      pb[nl * 40 + 16 + col] = (_Float16)(e1[j] * msk[ro + mB]);
    }
    v16h aP = load_a16(&smem[wv][col * 40], hi8);
    // --- PV: two WMMAs (K = m-chunk = 32, N = d halves) ---
    const _Float16* vb0 = Vt + ((size_t)bh * DH_ + col) * L_ + m0 + (hi << 4);
    v16h bV0 = *(const v16h*)vb0;
    v16h bV1 = *(const v16h*)(vb0 + 16 * (size_t)L_);
    o0 = wmma16(aP, bV0, o0);
    o1 = wmma16(aP, bV1, o1);
  }
  // --- normalize, leaky-ReLU, store [ROWS,320] f16 ---
#pragma unroll
  for (int j = 0; j < 8; ++j) {
    size_t grow = (size_t)b * L_ + ntile * 16 + hi8 + j;
    float inv = 1.0f / rsum[j];
    float v0 = o0[j] * inv, v1 = o1[j] * inv;
    v0 = v0 > 0.f ? v0 : 0.01f * v0;
    v1 = v1 > 0.f ? v1 : 0.01f * v1;
    attn[grow * DM_ + h * DH_ + col]      = (_Float16)v0;
    attn[grow * DM_ + h * DH_ + 16 + col] = (_Float16)v1;
  }
}

// ---------------------------------------------------------------------------
// Out-projection + residual: Zres = attn@w_o + Z   (f32 out)
// ---------------------------------------------------------------------------
__global__ void __launch_bounds__(256)
wo_gemm_kernel(const _Float16* __restrict__ A, const _Float16* __restrict__ Bpk,
               const float* __restrict__ Zin, float* __restrict__ Zres) {
  const int NT = 20, KT = 10;
  int gw   = blockIdx.x * (blockDim.x >> 5) + (threadIdx.x >> 5);
  int lane = threadIdx.x & 31;
  if (gw >= (ROWS_ / 16) * NT) return;
  int mt = gw / NT, nt = gw % NT;
  v8f acc = gemm_tile(A, DM_, Bpk, mt, nt, NT, KT, lane);
  int hi8 = ((lane >> 4) << 3), col = lane & 15;
  int cg = nt * 16 + col;
#pragma unroll
  for (int j = 0; j < 8; ++j) {
    size_t row = (size_t)(mt * 16 + hi8 + j);
    Zres[row * DM_ + cg] = acc[j] + Zin[row * DM_ + cg];
  }
}

// FFN1: H = relu(Zn2@w1 + b1) -> f16
__global__ void __launch_bounds__(256)
ffn1_gemm_kernel(const _Float16* __restrict__ A, const _Float16* __restrict__ Bpk,
                 const float* __restrict__ b1, _Float16* __restrict__ H) {
  const int NT = 80, KT = 10;
  int gw   = blockIdx.x * (blockDim.x >> 5) + (threadIdx.x >> 5);
  int lane = threadIdx.x & 31;
  if (gw >= (ROWS_ / 16) * NT) return;
  int mt = gw / NT, nt = gw % NT;
  v8f acc = gemm_tile(A, DM_, Bpk, mt, nt, NT, KT, lane);
  int hi8 = ((lane >> 4) << 3), col = lane & 15;
  int cg = nt * 16 + col;
  float bv = b1[cg];
#pragma unroll
  for (int j = 0; j < 8; ++j) {
    size_t row = (size_t)(mt * 16 + hi8 + j);
    float v = acc[j] + bv;
    H[row * DI_ + cg] = (_Float16)(v > 0.f ? v : 0.f);
  }
}

// FFN2: out = H@w2 + b2 + Zres  (f32)
__global__ void __launch_bounds__(256)
ffn2_gemm_kernel(const _Float16* __restrict__ A, const _Float16* __restrict__ Bpk,
                 const float* __restrict__ b2, const float* __restrict__ Zres,
                 float* __restrict__ out) {
  const int NT = 20, KT = 40;
  int gw   = blockIdx.x * (blockDim.x >> 5) + (threadIdx.x >> 5);
  int lane = threadIdx.x & 31;
  if (gw >= (ROWS_ / 16) * NT) return;
  int mt = gw / NT, nt = gw % NT;
  v8f acc = gemm_tile(A, DI_, Bpk, mt, nt, NT, KT, lane);
  int hi8 = ((lane >> 4) << 3), col = lane & 15;
  int cg = nt * 16 + col;
  float bv = b2[cg];
#pragma unroll
  for (int j = 0; j < 8; ++j) {
    size_t row = (size_t)(mt * 16 + hi8 + j);
    out[row * DM_ + cg] = acc[j] + bv + Zres[row * DM_ + cg];
  }
}

// ---------------------------------------------------------------------------
extern "C" void kernel_launch(void* const* d_in, const int* in_sizes, int n_in,
                              void* d_out, int out_size, void* d_ws,
                              size_t ws_size, hipStream_t stream) {
  (void)in_sizes; (void)n_in; (void)out_size; (void)ws_size;
  const float* Z    = (const float*)d_in[0];
  const float* Dm   = (const float*)d_in[1];
  const float* nmsk = (const float*)d_in[2];
  const float* msk  = (const float*)d_in[3];
  const float* gam  = (const float*)d_in[4];
  const float* wqkv = (const float*)d_in[5];
  const float* bqkv = (const float*)d_in[6];
  const float* wo   = (const float*)d_in[7];
  const float* ln1g = (const float*)d_in[8];
  const float* ln1b = (const float*)d_in[9];
  const float* ln2g = (const float*)d_in[10];
  const float* ln2b = (const float*)d_in[11];
  const float* w1   = (const float*)d_in[12];
  const float* b1   = (const float*)d_in[13];
  const float* w2   = (const float*)d_in[14];
  const float* b2   = (const float*)d_in[15];
  const int*   pN   = (const int*)d_in[16];
  const int*   pM   = (const int*)d_in[17];
  float* out = (float*)d_out;

  char* ws = (char*)d_ws;
  size_t off = 0;
  auto take = [&](size_t bytes) -> char* {
    char* p = ws + off;
    off = (off + bytes + 255) & ~(size_t)255;
    return p;
  };
  _Float16* zn1    = (_Float16*)take((size_t)ROWS_ * DM_ * 2);
  _Float16* wqkvpk = (_Float16*)take((size_t)DM_ * 3 * DM_ * 2);
  _Float16* wopk   = (_Float16*)take((size_t)DM_ * DM_ * 2);
  _Float16* w1pk   = (_Float16*)take((size_t)DM_ * DI_ * 2);
  _Float16* w2pk   = (_Float16*)take((size_t)DI_ * DM_ * 2);
  _Float16* Qh     = (_Float16*)take((size_t)B_ * NH_ * L_ * DH_ * 2);
  _Float16* Kh     = (_Float16*)take((size_t)B_ * NH_ * L_ * DH_ * 2);
  _Float16* Vt     = (_Float16*)take((size_t)B_ * NH_ * L_ * DH_ * 2);
  _Float16* attn   = (_Float16*)take((size_t)ROWS_ * DM_ * 2);
  float*    Zres   = (float*)take((size_t)ROWS_ * DM_ * 4);
  _Float16* zn2    = (_Float16*)take((size_t)ROWS_ * DM_ * 2);
  _Float16* Hbuf   = (_Float16*)take((size_t)ROWS_ * DI_ * 2);

  // 1) LN1
  ln_f16_kernel<<<ROWS_ / 8, 256, 0, stream>>>(Z, ln1g, ln1b, zn1);
  // 2) pack weights into B-fragment layout
  pack_b_kernel<<<(DM_ * 3 * DM_ + 255) / 256, 256, 0, stream>>>(wqkv, wqkvpk, DM_, 3 * DM_);
  pack_b_kernel<<<(DM_ * DM_ + 255) / 256, 256, 0, stream>>>(wo, wopk, DM_, DM_);
  pack_b_kernel<<<(DM_ * DI_ + 255) / 256, 256, 0, stream>>>(w1, w1pk, DM_, DI_);
  pack_b_kernel<<<(DI_ * DM_ + 255) / 256, 256, 0, stream>>>(w2, w2pk, DI_, DM_);
  // 3) QKV gemm + scatter
  qkv_gemm_kernel<<<(ROWS_ / 16) * 60 / 8, 256, 0, stream>>>(zn1, wqkvpk, bqkv, Qh, Kh, Vt);
  // 4) fused flash attention (one wave per (b,h,16-row tile))
  attn_kernel<<<B_ * NH_ * (L_ / 16) / 4, 128, 0, stream>>>(Qh, Kh, Vt, Dm, nmsk,
                                                            msk, gam, pN, pM, attn);
  // 5) out-proj + residual
  wo_gemm_kernel<<<(ROWS_ / 16) * 20 / 8, 256, 0, stream>>>(attn, wopk, Z, Zres);
  // 6) LN2
  ln_f16_kernel<<<ROWS_ / 8, 256, 0, stream>>>(Zres, ln2g, ln2b, zn2);
  // 7) FFN1 (relu)
  ffn1_gemm_kernel<<<(ROWS_ / 16) * 80 / 8, 256, 0, stream>>>(zn2, w1pk, b1, Hbuf);
  // 8) FFN2 + bias + residual
  ffn2_gemm_kernel<<<(ROWS_ / 16) * 20 / 8, 256, 0, stream>>>(Hbuf, w2pk, b2, Zres, out);
}
